// GraphSim_70824010711521
// MI455X (gfx1250) — matrix operand order repaired and checked by reading, not compiled, not run on hardware
//
#include <hip/hip_runtime.h>
#include <math.h>

// ---------------- CDNA5 WMMA types ----------------
typedef __attribute__((ext_vector_type(16))) __bf16 v16bf;
typedef __attribute__((ext_vector_type(8)))  float  v8f;

#define NN      65536   // nodes per side (B*NPG)
#define BGR     128     // graph pairs
#define NPGC    512     // nodes per graph
#define RD      10      // resize dim
#define CCHN    8       // conv channels
#define FEATLEN 2400    // 3*8*10*10

__device__ __forceinline__ unsigned short bf16_bits(float f) {
  union { float f; unsigned u; } un; un.f = f;
  unsigned r = un.u + 0x7fffu + ((un.u >> 16) & 1u);   // round-to-nearest-even
  return (unsigned short)(r >> 16);
}
__device__ __forceinline__ __bf16 f2bf(float f) {
  union { unsigned short s; __bf16 b; } ub;
  ub.s = bf16_bits(f);
  return ub.b;
}

// ---------------- utility kernels ----------------
__global__ void zero_f_kernel(float* p, long n) {
  long i = (long)blockIdx.x * blockDim.x + threadIdx.x;
  if (i < n) p[i] = 0.0f;
}
__global__ void zero_i_kernel(int* p, long n) {
  long i = (long)blockIdx.x * blockDim.x + threadIdx.x;
  if (i < n) p[i] = 0;
}
__global__ void deg_kernel(const int* __restrict__ dst, int* __restrict__ deg, int E) {
  int i = blockIdx.x * blockDim.x + threadIdx.x;
  if (i < E) atomicAdd(&deg[dst[i]], 1);
}
__global__ void inv_kernel(const int* __restrict__ deg, float* __restrict__ inv, int n) {
  int i = blockIdx.x * blockDim.x + threadIdx.x;
  if (i < n) inv[i] = rsqrtf((float)deg[i] + 1.0f);
}

// float -> packed bf16 (2 elements / thread, dword stores)
__global__ void cvt_bf16_kernel(const float* __restrict__ X, unsigned* __restrict__ Xb, long n2) {
  long i = (long)blockIdx.x * blockDim.x + threadIdx.x;
  if (i >= n2) return;
  unsigned lo = bf16_bits(X[2 * i]), hi = bf16_bits(X[2 * i + 1]);
  Xb[i] = lo | (hi << 16);
}

// Pack W[K,Nc] into exact WMMA B-fragment order:
// Wp[((nt*(K/32)+ks)*32+lane)*8 + v] = {W[k,n], W[k+1,n]} bf16-packed,
// n = nt*16 + lane%16, k = ks*32 + (lane<16?0:16) + 2v.
__global__ void pack_w_kernel(const float* __restrict__ W, unsigned* __restrict__ Wp,
                              int K, int Nc) {
  int t = blockIdx.x * blockDim.x + threadIdx.x;
  int total = (Nc >> 4) * (K >> 5) * 32 * 8;
  if (t >= total) return;
  int v = t & 7;
  int lane = (t >> 3) & 31;
  int ks = (t >> 8) % (K >> 5);
  int nt = (t >> 8) / (K >> 5);
  int n = (nt << 4) + (lane & 15);
  int k = (ks << 5) + ((lane < 16) ? 0 : 16) + 2 * v;
  unsigned lo = bf16_bits(W[(size_t)k * Nc + n]);
  unsigned hi = bf16_bits(W[(size_t)(k + 1) * Nc + n]);
  Wp[t] = lo | (hi << 16);
}

// ---------------- GEMM: XW[NN,NT*16] = Xb[NN,K](bf16) @ Wp (packed bf16) --------
// One wave computes a 16 x (NT*16) strip: A fragment reused across NT WMMAs.
// Inner loop: 2x b128 (A) + NT * 2x b128 (B) + NT * v_wmma. No conversions.
template <int K, int NT>
__global__ void gemm_bf16_wmma(const unsigned short* __restrict__ Xb,
                               const unsigned* __restrict__ Wp,
                               float* __restrict__ XW) {
  constexpr int Nc = NT * 16;
  const int wave = threadIdx.x >> 5;
  const int lane = threadIdx.x & 31;
  const int tileM = blockIdx.x * 4 + wave;
  const int row = (tileM << 4) + (lane & 15);
  const int kA = (lane < 16) ? 0 : 16;    // A 16x32 bf16 layout: contiguous k = 16h + s
  v8f c[NT];
#pragma unroll
  for (int nt = 0; nt < NT; ++nt) c[nt] = (v8f){0.f,0.f,0.f,0.f,0.f,0.f,0.f,0.f};
#pragma unroll
  for (int ks = 0; ks < K / 32; ++ks) {
    const int kk = ks * 32;
    union { v16bf v; uint4 u[2]; } A;
    const uint4* ap = reinterpret_cast<const uint4*>(Xb + (size_t)row * K + kk + kA);
    A.u[0] = ap[0];   // K = kk+kA .. +7
    A.u[1] = ap[1];   // K = kk+kA+8 .. +7
#pragma unroll
    for (int nt = 0; nt < NT; ++nt) {
      union { v16bf v; uint4 u[2]; } B;
      const uint4* bp = reinterpret_cast<const uint4*>(
          Wp + ((size_t)((nt * (K / 32) + ks) * 32 + lane)) * 8);
      B.u[0] = bp[0];
      B.u[1] = bp[1];
      c[nt] = __builtin_amdgcn_wmma_f32_16x16x32_bf16(false, A.v, false, B.v,
                                                      (short)0, c[nt], false, false);
    }
  }
  const int mo = (lane < 16) ? 0 : 8;
  const int col0 = lane & 15;
#pragma unroll
  for (int nt = 0; nt < NT; ++nt)
#pragma unroll
    for (int r = 0; r < 8; ++r)
      XW[(size_t)((tileM << 4) + r + mo) * Nc + nt * 16 + col0] = c[nt][r];
}

// ---------------- GCN edge aggregation (atomic scatter-add) ----------------
// FO compile-time -> index math is shifts/masks; fo threads per edge, coalesced
// row gather + atomic row scatter; scalar edge/inv loads broadcast within group.
template <int FO>
__global__ void agg_kernel(const int* __restrict__ src, const int* __restrict__ dst,
                           const float* __restrict__ xw, const float* __restrict__ inv,
                           float* __restrict__ agg, int E) {
  long t = (long)blockIdx.x * blockDim.x + threadIdx.x;
  if (t >= (long)E * FO) return;
  int e = (int)(t / FO), f = (int)(t & (FO - 1));
  int s = src[e], d = dst[e];
  atomicAdd(&agg[(size_t)d * FO + f], xw[(size_t)s * FO + f] * inv[s] * inv[d]);
}

// out = agg + xw*inv^2 + b (+ReLU); also emits bf16 copy for next GEMM.
// 2 consecutive elements per thread (FO even -> same node, f even).
template <int FO, int RELU>
__global__ void combine_kernel(const float* __restrict__ agg, const float* __restrict__ xw,
                               const float* __restrict__ inv, const float* __restrict__ bias,
                               float* __restrict__ outf, unsigned* __restrict__ outb) {
  long i2 = (long)blockIdx.x * blockDim.x + threadIdx.x;
  if (i2 >= (long)NN * FO / 2) return;
  long i = 2 * i2;
  int node = (int)(i / FO), f = (int)(i & (FO - 1));
  float iv2 = inv[node] * inv[node];
  float v0 = agg[i] + xw[i] * iv2 + bias[f];
  float v1 = agg[i + 1] + xw[i + 1] * iv2 + bias[f + 1];
  if (RELU) { v0 = fmaxf(v0, 0.f); v1 = fmaxf(v1, 0.f); }
  outf[i] = v0;
  outf[i + 1] = v1;
  outb[i2] = (unsigned)bf16_bits(v0) | ((unsigned)bf16_bits(v1) << 16);
}

// ---------------- bilinear pooling: p[b,u,:] = w0*h[b,r0,:] + w1*h[b,r0+1,:] ----
// resize coord = (u+0.5)*(512/10)-0.5, always interior (no clamp).
template <int FO>
__global__ void pool_kernel(const float* __restrict__ h, float* __restrict__ p) {
  int t = blockIdx.x * blockDim.x + threadIdx.x;   // over BGR*RD*FO
  int f = t & (FO - 1);
  int u = (t / FO) % RD;
  int b = t / (FO * RD);
  float cc = (u + 0.5f) * ((float)NPGC / (float)RD) - 0.5f;
  int r0 = (int)floorf(cc);
  float w1 = cc - (float)r0, w0 = 1.0f - w1;
  p[((size_t)b * RD + u) * FO + f] =
      w0 * h[((size_t)b * NPGC + r0) * FO + f] + w1 * h[((size_t)b * NPGC + r0 + 1) * FO + f];
}

// ---------------- per-graph simr(10x10 in 16x16 WMMA tile) + 3x3 conv + ReLU ----
template <int FO>
__global__ void sim_conv_kernel(const float* __restrict__ PI_, const float* __restrict__ PJ_,
                                const float* __restrict__ cw, const float* __restrict__ cb,
                                float* __restrict__ feat, int li) {
  __shared__ float s[4][12][12];
  const int wave = threadIdx.x >> 5, lane = threadIdx.x & 31;
  const int b = blockIdx.x * 4 + wave;
  for (int idx = lane; idx < 144; idx += 32) s[wave][idx / 12][idx % 12] = 0.0f;
  __syncthreads();

  const float* pi = PI_ + (size_t)b * RD * FO;
  const float* pj = PJ_ + (size_t)b * RD * FO;
  const int m  = lane & 15;
  const int kA = (lane < 16) ? 0 : 16;
  const int kB = (lane < 16) ? 0 : 16;
  constexpr int KK = (FO < 32) ? 32 : FO;
  v8f c = {0.f,0.f,0.f,0.f,0.f,0.f,0.f,0.f};
#pragma unroll
  for (int kk = 0; kk < KK; kk += 32) {
    v16bf a, bb;
#pragma unroll
    for (int v = 0; v < 4; ++v) {
      int k0 = kk + kA + 2 * v;
      a[2*v]     = (m < RD && k0     < FO) ? f2bf(pi[m * FO + k0])     : f2bf(0.0f);
      a[2*v + 1] = (m < RD && k0 + 1 < FO) ? f2bf(pi[m * FO + k0 + 1]) : f2bf(0.0f);
      int k1 = kk + 8 + kA + 2 * v;
      a[8 + 2*v]     = (m < RD && k1     < FO) ? f2bf(pi[m * FO + k1])     : f2bf(0.0f);
      a[8 + 2*v + 1] = (m < RD && k1 + 1 < FO) ? f2bf(pi[m * FO + k1 + 1]) : f2bf(0.0f);
    }
#pragma unroll
    for (int v = 0; v < 8; ++v) {
      int k = kk + kB + 2 * v;
      bb[2*v]     = (m < RD && k     < FO) ? f2bf(pj[m * FO + k])     : f2bf(0.0f);
      bb[2*v + 1] = (m < RD && k + 1 < FO) ? f2bf(pj[m * FO + k + 1]) : f2bf(0.0f);
    }
    c = __builtin_amdgcn_wmma_f32_16x16x32_bf16(false, a, false, bb, (short)0, c, false, false);
  }
  const int mo = (lane < 16) ? 0 : 8;
#pragma unroll
  for (int r = 0; r < 8; ++r) {
    int M = r + mo;
    if (M < RD && m < RD) s[wave][M + 1][m + 1] = c[r];
  }
  __syncthreads();

  for (int o = lane; o < CCHN * RD * RD; o += 32) {
    int ch = o / (RD * RD); int rem = o % (RD * RD);
    int u = rem / RD, v = rem % RD;
    float acc = cb[ch];
#pragma unroll
    for (int du = 0; du < 3; ++du)
#pragma unroll
      for (int dv = 0; dv < 3; ++dv)
        acc += cw[ch * 9 + du * 3 + dv] * s[wave][u + du][v + dv];
    feat[(size_t)b * FEATLEN + li * (CCHN * RD * RD) + o] = fmaxf(acc, 0.0f);
  }
}

// ---------------- MLP 2400->32->16->8->4->1 + score + sigmoid ----------------
// 256 threads/graph; layer0 split 8 ways (serial length 300) + LDS reduction.
__global__ void mlp_kernel(const float* __restrict__ feat,
                           const float* w0, const float* b0, const float* w1, const float* b1,
                           const float* w2, const float* b2, const float* w3, const float* b3,
                           const float* w4, const float* b4,
                           const float* sw, const float* sb, float* __restrict__ out) {
  __shared__ float part[8][32];
  __shared__ float h0[32], h1[16], h2[8], h3[4];
  int b = blockIdx.x, t = threadIdx.x;
  const float* f = feat + (size_t)b * FEATLEN;
  int o = t & 31, chunk = t >> 5;                 // 8 chunks of 300
  float s = 0.0f;
  for (int k = chunk * 300; k < (chunk + 1) * 300; ++k) s += f[k] * w0[k * 32 + o];
  part[chunk][o] = s;
  __syncthreads();
  if (t < 32) {
    float acc = b0[t];
#pragma unroll
    for (int cidx = 0; cidx < 8; ++cidx) acc += part[cidx][t];
    h0[t] = fmaxf(acc, 0.0f);
  }
  __syncthreads();
  if (t < 16) {
    float a = b1[t];
    for (int k = 0; k < 32; ++k) a += h0[k] * w1[k * 16 + t];
    h1[t] = fmaxf(a, 0.0f);
  }
  __syncthreads();
  if (t < 8) {
    float a = b2[t];
    for (int k = 0; k < 16; ++k) a += h1[k] * w2[k * 8 + t];
    h2[t] = fmaxf(a, 0.0f);
  }
  __syncthreads();
  if (t < 4) {
    float a = b3[t];
    for (int k = 0; k < 8; ++k) a += h2[k] * w3[k * 4 + t];
    h3[t] = fmaxf(a, 0.0f);
  }
  __syncthreads();
  if (t == 0) {
    float a = b4[0];
    for (int k = 0; k < 4; ++k) a += h3[k] * w4[k];
    a = fmaxf(a, 0.0f);
    float sc = a * sw[0] + sb[0];
    out[b] = 1.0f / (1.0f + expf(-sc));
  }
}

// ---------------- host orchestration ----------------
extern "C" void kernel_launch(void* const* d_in, const int* in_sizes, int n_in,
                              void* d_out, int out_size, void* d_ws, size_t ws_size,
                              hipStream_t stream) {
  (void)n_in; (void)out_size; (void)ws_size;
  const float* x0[2]  = {(const float*)d_in[0], (const float*)d_in[1]};
  const float* gw[3]  = {(const float*)d_in[2], (const float*)d_in[4], (const float*)d_in[6]};
  const float* gb[3]  = {(const float*)d_in[3], (const float*)d_in[5], (const float*)d_in[7]};
  const float* cwp[3] = {(const float*)d_in[8], (const float*)d_in[10], (const float*)d_in[12]};
  const float* cbp[3] = {(const float*)d_in[9], (const float*)d_in[11], (const float*)d_in[13]};
  const int* eidx[2]  = {(const int*)d_in[26], (const int*)d_in[27]};
  const int E = in_sizes[26] / 2;

  // workspace carve (floats) — ~86 MB total; every sub-buffer 256B-aligned
  float* ws = (float*)d_ws;
  size_t off = 0;
  float* INV[2]; INV[0] = ws + off; off += NN; INV[1] = ws + off; off += NN;
  float* XW  = ws + off; off += (size_t)NN * 64;
  float* AGG = ws + off; off += (size_t)NN * 64;
  float* ACT[2];  ACT[0]  = ws + off; off += (size_t)NN * 64; ACT[1]  = ws + off; off += (size_t)NN * 64;
  unsigned short* ACTB[2];
  ACTB[0] = (unsigned short*)(ws + off); off += (size_t)NN * 32;   // NN*64 bf16
  ACTB[1] = (unsigned short*)(ws + off); off += (size_t)NN * 32;
  float* PP[2];  PP[0]  = ws + off; off += (size_t)BGR * RD * 64; PP[1] = ws + off; off += (size_t)BGR * RD * 64;
  float* FEAT = ws + off; off += (size_t)BGR * FEATLEN;
  unsigned* WPACK = (unsigned*)(ws + off); off += 2048;
  int* DEG = (int*)(ws + off); off += NN;

  // normalization coefficients (layer-independent, once per side)
  for (int side = 0; side < 2; ++side) {
    zero_i_kernel<<<NN / 256, 256, 0, stream>>>(DEG, NN);
    deg_kernel<<<(E + 255) / 256, 256, 0, stream>>>(eidx[side] + E, DEG, E);
    inv_kernel<<<NN / 256, 256, 0, stream>>>(DEG, INV[side], NN);
    // layer-0 GEMM input in packed bf16
    cvt_bf16_kernel<<<(NN * 64 / 2 + 255) / 256, 256, 0, stream>>>(
        x0[side], (unsigned*)ACTB[side], (long)NN * 32);
  }

  const int fi_arr[3] = {64, 64, 32};
  const int fo_arr[3] = {64, 32, 16};
  for (int li = 0; li < 3; ++li) {
    const int fi = fi_arr[li], fo = fo_arr[li];
    // pack weights into B-fragment order (shared by both sides)
    int wtot = (fo / 16) * (fi / 32) * 256;
    pack_w_kernel<<<(wtot + 255) / 256, 256, 0, stream>>>(gw[li], WPACK, fi, fo);
    for (int side = 0; side < 2; ++side) {
      long nfo = (long)NN * fo;
      long efo = (long)E * fo;
      unsigned gb_agg = (unsigned)((efo + 255) / 256);
      unsigned gb_cmb = (unsigned)((nfo / 2 + 255) / 256);
      const int* esrc = eidx[side];
      const int* edst = eidx[side] + E;
      zero_f_kernel<<<(unsigned)((nfo + 255) / 256), 256, 0, stream>>>(AGG, nfo);
      if (li == 0) {
        gemm_bf16_wmma<64, 4><<<NN / 64, 128, 0, stream>>>(ACTB[side], WPACK, XW);
        agg_kernel<64><<<gb_agg, 256, 0, stream>>>(esrc, edst, XW, INV[side], AGG, E);
        combine_kernel<64, 1><<<gb_cmb, 256, 0, stream>>>(AGG, XW, INV[side], gb[li],
                                                          ACT[side], (unsigned*)ACTB[side]);
        pool_kernel<64><<<BGR * RD * 64 / 256, 256, 0, stream>>>(ACT[side], PP[side]);
      } else if (li == 1) {
        gemm_bf16_wmma<64, 2><<<NN / 64, 128, 0, stream>>>(ACTB[side], WPACK, XW);
        agg_kernel<32><<<gb_agg, 256, 0, stream>>>(esrc, edst, XW, INV[side], AGG, E);
        combine_kernel<32, 1><<<gb_cmb, 256, 0, stream>>>(AGG, XW, INV[side], gb[li],
                                                          ACT[side], (unsigned*)ACTB[side]);
        pool_kernel<32><<<BGR * RD * 32 / 256, 256, 0, stream>>>(ACT[side], PP[side]);
      } else {
        gemm_bf16_wmma<32, 1><<<NN / 64, 128, 0, stream>>>(ACTB[side], WPACK, XW);
        agg_kernel<16><<<gb_agg, 256, 0, stream>>>(esrc, edst, XW, INV[side], AGG, E);
        combine_kernel<16, 0><<<gb_cmb, 256, 0, stream>>>(AGG, XW, INV[side], gb[li],
                                                          ACT[side], (unsigned*)ACTB[side]);
        pool_kernel<16><<<BGR * RD * 16 / 256, 256, 0, stream>>>(ACT[side], PP[side]);
      }
    }
    if (li == 0)      sim_conv_kernel<64><<<BGR / 4, 128, 0, stream>>>(PP[0], PP[1], cwp[0], cbp[0], FEAT, 0);
    else if (li == 1) sim_conv_kernel<32><<<BGR / 4, 128, 0, stream>>>(PP[0], PP[1], cwp[1], cbp[1], FEAT, 1);
    else              sim_conv_kernel<16><<<BGR / 4, 128, 0, stream>>>(PP[0], PP[1], cwp[2], cbp[2], FEAT, 2);
  }

  mlp_kernel<<<BGR, 256, 0, stream>>>(
      FEAT,
      (const float*)d_in[14], (const float*)d_in[15], (const float*)d_in[16], (const float*)d_in[17],
      (const float*)d_in[18], (const float*)d_in[19], (const float*)d_in[20], (const float*)d_in[21],
      (const float*)d_in[22], (const float*)d_in[23], (const float*)d_in[24], (const float*)d_in[25],
      (float*)d_out);
}
